// LePEAttention_63814624084283
// MI455X (gfx1250) — matrix operand order repaired and checked by reading, not compiled, not run on hardware
//
#include <hip/hip_runtime.h>

typedef __attribute__((ext_vector_type(16))) _Float16 v16h;
typedef __attribute__((ext_vector_type(8)))  _Float16 v8h;
typedef __attribute__((ext_vector_type(8)))  float    v8f;
typedef __attribute__((ext_vector_type(4)))  float    v4f;

// Static problem configuration (matches reference)
constexpr int Bn  = 2;
constexpr int T   = 8;
constexpr int Hh  = 64;
constexpr int Ww  = 64;
constexpr int C   = 128;
constexpr int TSP = 2, HSP = 64, WSP = 4;
constexpr int NH  = 4;
constexpr int HD  = 32;                 // head dim
constexpr int L   = T * Hh * Ww;        // 32768
constexpr int N   = TSP * HSP * WSP;    // 512 tokens per window
constexpr int NWB = (T/TSP) * (Hh/HSP) * (Ww/WSP);  // 64 windows per batch
constexpr int NW  = Bn * NWB;           // 128 windows
// q scale folded with log2(e) so softmax uses exp2 directly (v_exp_f32, no mul)
#define QSCALE_LOG2E 0.2550404171817875f   // 32^-0.5 * 1.4426950408889634

// raw v_exp_f32: flush-to-zero below -126 is exactly right for softmax tails
__device__ __forceinline__ float fast_exp2(float x) {
#if __has_builtin(__builtin_amdgcn_exp2f)
  return __builtin_amdgcn_exp2f(x);
#else
  return exp2f(x);
#endif
}

// After call: a = hi_half ? partner(b) : a ; b = hi_half ? b : partner(a)
// (i.e. v_permlane16_swap_b32 semantics: a[16..31] <-> b[0..15])
__device__ __forceinline__ void half_swap(float& a, float& b) {
#if __has_builtin(__builtin_amdgcn_permlane16_swap)
  typedef unsigned int v2u __attribute__((ext_vector_type(2)));
  v2u r = __builtin_amdgcn_permlane16_swap(__builtin_bit_cast(unsigned int, a),
                                           __builtin_bit_cast(unsigned int, b),
                                           false, false);
  a = __builtin_bit_cast(float, r.x);
  b = __builtin_bit_cast(float, r.y);
#else
  const bool hi = (threadIdx.x & 16) != 0;
  const float oa = __shfl_xor(a, 16, 32);
  const float ob = __shfl_xor(b, 16, 32);
  const float na = hi ? ob : a;
  const float nb = hi ? b : oa;
  a = na; b = nb;
#endif
}

// token index inside window -> flat l in (T,H,W); hB==0 since H/HSP==1
__device__ __forceinline__ int token_l(int tB, int wB, int n) {
  int t = n >> 8;           // n = t*256 + h*4 + w
  int h = (n >> 2) & 63;
  int w = n & 3;
  return ((tB * TSP + t) * Hh + h) * Ww + (wB * WSP + w);
}

__global__ __launch_bounds__(512, 1)
void lepe_attn_kernel(const float* __restrict__ qkv,
                      const float* __restrict__ lepe_w,
                      const float* __restrict__ lepe_b,
                      float* __restrict__ out)
{
  __shared__ _Float16 sK [N * HD];        // [n][d]  K row-major (WMMA A), 32 KB
  __shared__ _Float16 sVt[HD * N];        // [d][n]  V transposed  (WMMA A), 32 KB
  __shared__ _Float16 sV [N * HD];        // [n][d]  V row-major (LePE conv), 32 KB
  __shared__ float    sWt[28 * HD];       // [tap][d] conv weights; row 27 = zeros
  __shared__ float    sB [HD];            // bias for this head

  const int blk  = blockIdx.x;            // 0..511 = (window, head)
  const int head = blk & (NH - 1);
  const int win  = blk >> 2;
  const int b    = win / NWB;
  const int wib  = win % NWB;             // tB*16 + wB
  const int tB   = wib >> 4;
  const int wB   = wib & 15;

  const int tid  = threadIdx.x;
  const int lane = tid & 31;
  const int wave = tid >> 5;
  const bool hi  = lane >= 16;
  const int l16  = lane & 15;

  const float* qg = qkv;
  const float* kg = qkv + (size_t)Bn * L * C;
  const float* vg = qkv + (size_t)2 * Bn * L * C;
  const int cbase = head * HD;

  // ---- stage K, V, V^T (f16) into LDS; one token/thread ----
  {
    const int n = tid;
    const int l = token_l(tB, wB, n);
    const float* kp = kg + ((size_t)b * L + l) * C + cbase;
    const float* vp = vg + ((size_t)b * L + l) * C + cbase;
#pragma unroll
    for (int d0 = 0; d0 < HD; d0 += 4) {
      v4f kk = *reinterpret_cast<const v4f*>(kp + d0);
      v4f vv = *reinterpret_cast<const v4f*>(vp + d0);
#pragma unroll
      for (int j = 0; j < 4; ++j) {
        sK [n * HD + d0 + j]  = (_Float16)kk[j];
        sV [n * HD + d0 + j]  = (_Float16)vv[j];
        sVt[(d0 + j) * N + n] = (_Float16)vv[j];
      }
    }
  }
  // ---- stage LePE weights transposed [tap][d] (+ zero row 27) and bias ----
  for (int i = tid; i < 28 * HD; i += 512) {
    const int tap = i >> 5;          // i / 32
    const int d   = i & 31;
    sWt[i] = (tap < 27) ? lepe_w[(size_t)(cbase + d) * 27 + tap] : 0.f;
  }
  if (tid < HD) sB[tid] = lepe_b[cbase + tid];
  __syncthreads();

  // 16 waves sweep the 32 query tiles
#pragma unroll 1
  for (int qt = wave; qt < N / 16; qt += 16) {
    const int qbase = qt * 16;
    const int qn = qbase + l16;            // this lane's query token
    const int ql = token_l(tB, wB, qn);

    // prefetch next query tile for this wave
    if (qt + 16 < N / 16) {
      const int pl = token_l(tB, wB, (qt + 16) * 16 + l16);
      __builtin_prefetch(qg + ((size_t)b * L + pl) * C + cbase, 0, 1);
    }

    // B operand: Q^T (32d x 16q). lane = query col; lanes0-15 d0..7,16..23, lanes16-31 d8..15,24..31
    v16h bq;
    {
      const float* qp = qg + ((size_t)b * L + ql) * C + cbase + (hi ? 8 : 0);
#pragma unroll
      for (int j = 0; j < 16; j += 4) {
        v4f qq = *reinterpret_cast<const v4f*>(qp + j + (j & 8));
        bq[j + 0] = (_Float16)(qq[0] * QSCALE_LOG2E);
        bq[j + 1] = (_Float16)(qq[1] * QSCALE_LOG2E);
        bq[j + 2] = (_Float16)(qq[2] * QSCALE_LOG2E);
        bq[j + 3] = (_Float16)(qq[3] * QSCALE_LOG2E);
      }
    }

    v8f acc0 = {}, acc1 = {};              // O^T: d0..15 / d16..31 (lane = query)
    float m = -1e30f, lsum = 0.f;          // m in log2 domain; lsum half-local
    const int g0 = hi ? 8 : 0;             // A-layout half-wave group offset

    for (int kb = 0; kb < N; kb += 32) {
      // A operands: two 16-key K tiles (16k x 32d), ISA A layout groups
      v16h aK0, aK1;
      {
        const _Float16* p0 = sK + (kb + l16) * HD;
        const _Float16* p1 = sK + (kb + 16 + l16) * HD;
        v8h* a0 = reinterpret_cast<v8h*>(&aK0);
        v8h* a1 = reinterpret_cast<v8h*>(&aK1);
        a0[0] = *reinterpret_cast<const v8h*>(p0 + g0);
        a0[1] = *reinterpret_cast<const v8h*>(p0 + g0 + 16);
        a1[0] = *reinterpret_cast<const v8h*>(p1 + g0);
        a1[1] = *reinterpret_cast<const v8h*>(p1 + g0 + 16);
      }
      v8f zero = {};
      // S^T tiles (log2-scaled): rows = keys, cols = queries; K=32 = full head_dim
      v8f st0 = __builtin_amdgcn_wmma_f32_16x16x32_f16(false, aK0, false, bq, (short)0, zero, false, false);
      v8f st1 = __builtin_amdgcn_wmma_f32_16x16x32_f16(false, aK1, false, bq, (short)0, zero, false, false);

      // online softmax (base-2): per-lane max over 16 key regs + xor16
      float tmax = st0[0];
#pragma unroll
      for (int r = 0; r < 8; ++r) {
        tmax = fmaxf(tmax, st0[r]);
        tmax = fmaxf(tmax, st1[r]);
      }
      tmax = fmaxf(tmax, __shfl_xor(tmax, 16, 32));
      const float mnew = fmaxf(m, tmax);
      const float corr = fast_exp2(m - mnew);
      m = mnew;
      lsum *= corr;

      float p0v[8], p1v[8];
      float psum = 0.f;
#pragma unroll
      for (int r = 0; r < 8; ++r) {
        p0v[r] = fast_exp2(st0[r] - m);
        p1v[r] = fast_exp2(st1[r] - m);
        psum += p0v[r] + p1v[r];
        acc0[r] *= corr;
        acc1[r] *= corr;
      }
      lsum += psum;   // half-local; combined across halves once after the loop

      // Build B operand P^T (32k x 16q): one permlane16_swap per reg pair
      v16h bp;
#pragma unroll
      for (int r = 0; r < 8; ++r) {
        float a = p0v[r], bb = p1v[r];
        half_swap(a, bb);
        bp[r]     = (_Float16)a;
        bp[r + 8] = (_Float16)bb;
      }

      // A operands: V^T halves (16d x 32k) from LDS, 2x b128 each
      v16h aV0, aV1;
      {
        const int kg0 = kb + 2 * g0;
        const _Float16* pv0 = sVt + l16 * N;
        const _Float16* pv1 = sVt + (16 + l16) * N;
        v8h* a0 = reinterpret_cast<v8h*>(&aV0);
        v8h* a1 = reinterpret_cast<v8h*>(&aV1);
        a0[0] = *reinterpret_cast<const v8h*>(pv0 + kg0);
        a0[1] = *reinterpret_cast<const v8h*>(pv0 + kg0 + 8);
        a1[0] = *reinterpret_cast<const v8h*>(pv1 + kg0);
        a1[1] = *reinterpret_cast<const v8h*>(pv1 + kg0 + 8);
      }
      acc0 = __builtin_amdgcn_wmma_f32_16x16x32_f16(false, aV0, false, bp, (short)0, acc0, false, false);
      acc1 = __builtin_amdgcn_wmma_f32_16x16x32_f16(false, aV1, false, bp, (short)0, acc1, false, false);
    }

    // combine the two half-wave partial row sums once
    lsum += __shfl_xor(lsum, 16, 32);

    // ---- epilogue: normalize + fused depthwise 3x3x3 LePE, all from LDS ----
    const float inv = 1.f / lsum;
    const int tt = qn >> 8;
    const int th = (qn >> 2) & 63;
    const int tw = qn & 3;
    float* op = out + ((size_t)b * L + ql) * C + cbase;

#pragma unroll
    for (int g = 0; g < 2; ++g) {
      const int dstart = (g ? 16 : 0) + (hi ? 8 : 0);  // 8 contiguous d per lane
      float r8[8];
#pragma unroll
      for (int r = 0; r < 8; ++r)
        r8[r] = (g ? acc1[r] : acc0[r]) * inv + sB[dstart + r];

#pragma unroll
      for (int dt = 0; dt < 3; ++dt) {
        const int zt = tt + dt - 1;
#pragma unroll
        for (int dh = 0; dh < 3; ++dh) {
          const int zh = th + dh - 1;
#pragma unroll
          for (int dw = 0; dw < 3; ++dw) {
            const int zw = tw + dw - 1;
            const bool ok = ((unsigned)zt < (unsigned)TSP) &
                            ((unsigned)zh < (unsigned)HSP) &
                            ((unsigned)zw < (unsigned)WSP);
            const int nn = ok ? ((zt << 8) + (zh << 2) + zw) : 0;
            const int wr = ok ? (dt * 9 + dh * 3 + dw) : 27;   // row 27 = zeros
            v8h vv = *reinterpret_cast<const v8h*>(sV + nn * HD + dstart);
            v4f w0 = *reinterpret_cast<const v4f*>(sWt + wr * HD + dstart);
            v4f w1 = *reinterpret_cast<const v4f*>(sWt + wr * HD + dstart + 4);
            r8[0] += w0[0] * (float)vv[0];
            r8[1] += w0[1] * (float)vv[1];
            r8[2] += w0[2] * (float)vv[2];
            r8[3] += w0[3] * (float)vv[3];
            r8[4] += w1[0] * (float)vv[4];
            r8[5] += w1[1] * (float)vv[5];
            r8[6] += w1[2] * (float)vv[6];
            r8[7] += w1[3] * (float)vv[7];
          }
        }
      }
      v4f o0 = { r8[0], r8[1], r8[2], r8[3] };
      v4f o1 = { r8[4], r8[5], r8[6], r8[7] };
      *reinterpret_cast<v4f*>(op + dstart)     = o0;
      *reinterpret_cast<v4f*>(op + dstart + 4) = o1;
    }
  }
}

extern "C" void kernel_launch(void* const* d_in, const int* in_sizes, int n_in,
                              void* d_out, int out_size, void* d_ws, size_t ws_size,
                              hipStream_t stream) {
  (void)in_sizes; (void)n_in; (void)out_size; (void)d_ws; (void)ws_size;
  const float* qkv = (const float*)d_in[0];
  const float* lw  = (const float*)d_in[1];
  const float* lb  = (const float*)d_in[2];
  float* out = (float*)d_out;

  dim3 grid(NW * NH);   // 512 blocks: one per (window, head)
  dim3 block(512);      // 16 wave32s
  hipLaunchKernelGGL(lepe_attn_kernel, grid, block, 0, stream, qkv, lw, lb, out);
}
